// RelationBlock_Out_70145405878465
// MI455X (gfx1250) — matrix operand order repaired and checked by reading, not compiled, not run on hardware
//
#include <hip/hip_runtime.h>
#include <cstdint>

// Problem constants
#define NV 4
#define NP 6
#define NB 8192
#define NF 128
#define BT 128          // batch tile rows per workgroup
#define NTILES 64       // NB / BT
#define PITCH 136       // LDS row pitch in halves (17 * 16B -> conflict-free b128)
#define WTILE (128 * PITCH)   // halves per LDS W buffer (34,816 B)

typedef _Float16 v16h __attribute__((ext_vector_type(16)));
typedef float    v8f  __attribute__((ext_vector_type(8)));
typedef unsigned int v4u __attribute__((ext_vector_type(4)));
typedef int          v8i __attribute__((ext_vector_type(8)));
typedef int          v4i __attribute__((ext_vector_type(4)));

union Frag16 { v16h v; uint4 q[2]; };

// Workspace layout (bytes). Total ~40.9 MB.
#define XH_OFF   0ul          // 4*8192*128 halves  = 8,388,608 B
#define WH_OFF   8388608ul    // 128*16384 halves   = 4,194,304 B
#define Y_OFF    12582912ul   // 6*8192*128 floats  = 25,165,824 B
#define S_OFF    37748736ul   // 6*128*512 floats   = 1,572,864 B
#define Q_OFF    39321600ul   // 6*128*512 floats   = 1,572,864 B
#define MEAN_OFF 40894464ul   // 768 floats
#define INV_OFF  40897536ul   // 768 floats

// ---------------------------------------------------------------- convert
__global__ void k_convert(const float* __restrict__ x, const float* __restrict__ W,
                          _Float16* __restrict__ xh, _Float16* __restrict__ Wh) {
  int idx = blockIdx.x * 256 + threadIdx.x;
  const int NX = NV * NB * NF;      // 4,194,304
  const int NW = NF * NF * NF;      // 2,097,152
  if (idx < NX) {
    xh[idx] = (_Float16)x[idx];
  } else if (idx < NX + NW) {
    int j = idx - NX;
    Wh[j] = (_Float16)W[j];
  }
}

// TDM: async-load one 128x128 f16 tile (row stride 128 halves in memory) into LDS,
// padding 4 DWORDs after every 64 DWORDs -> LDS row pitch = 136 halves (bank-safe).
// D# layout per CDNA5 ISA 8.3/8.4 (group0 128b, group1 256b; groups 2/3 unused: 2D).
__device__ __forceinline__ void tdm_load_tile(const _Float16* gsrc, unsigned lds_byte_off) {
  unsigned long long ga = (unsigned long long)(uintptr_t)gsrc;
  v4u g0;
  g0.x = 1u;                                   // count=1, is_restore=0, gather off
  g0.y = lds_byte_off;                         // lds_addr
  g0.z = (unsigned)ga;                         // global_addr[31:0]
  g0.w = (unsigned)(ga >> 32) | (2u << 30);    // global_addr[56:32] | type=2 ("image")
  v8i g1;
  g1[0] = (int)((1u << 16)    // data_size = 1 (2 bytes)
              | (1u << 20)    // pad_enable
              | (5u << 22)    // pad_interval: 64 DWORDs (one 128-half row)
              | (3u << 25));  // pad_amount: 4 DWORDs (8 halves)
  g1[1] = (int)(128u << 16);  // tensor_dim0 = 128 (low half in [31:16])
  g1[2] = (int)(128u << 16);  // tensor_dim0 hi=0 | tensor_dim1 = 128
  g1[3] = (int)(128u << 16);  // tensor_dim1 hi=0 | tile_dim0 = 128
  g1[4] = 128;                // tile_dim1 = 128, tile_dim2 = 0 (2D)
  g1[5] = 128;                // tensor_dim0_stride = 128
  g1[6] = 0;                  // stride hi / dim1_stride (unused, tile_dim2=0)
  g1[7] = 0;
  v4i z4 = {0, 0, 0, 0};
  v8i z8 = {0, 0, 0, 0, 0, 0, 0, 0};
  // 6-arg form (clang-23 / therock-10.0 headers): extra int32x8 before cpol.
  __builtin_amdgcn_tensor_load_to_lds(g0, g1, z4, z4, z8, 0);
}

// ---------------------------------------------------------------- main bilinear GEMM
__global__ void __launch_bounds__(256)
k_bilinear(const _Float16* __restrict__ xh, const _Float16* __restrict__ Wh,
           float* __restrict__ y, float* __restrict__ S, float* __restrict__ Q) {
  extern __shared__ _Float16 smdyn[];     // 2 * WTILE halves = 69,632 B (dynamic LDS)
  _Float16* const buf0 = smdyn;
  _Float16* const buf1 = smdyn + WTILE;

  const int tid  = threadIdx.x;
  const int bt   = blockIdx.x;     // batch tile 0..63
  const int p    = blockIdx.y;     // pair 0..5
  const int w    = tid >> 5;       // wave 0..7
  const int lane = tid & 31;
  const int ln   = lane & 15;
  const int hl   = lane >> 4;
  const int mb   = w * 16;         // wave's batch-row strip base inside tile
  const int b0   = bt * BT;

  const int II[NP] = {0, 0, 0, 1, 1, 2};
  const int JJ[NP] = {1, 2, 3, 2, 3, 3};
  const int v1 = II[p], v2 = JJ[p];

  // ---- stage X1 -> buf0, X2 -> buf1 (one pass) ----
  {
    const _Float16* s1 = xh + ((size_t)v1 * NB + b0) * NF;
    const _Float16* s2 = xh + ((size_t)v2 * NB + b0) * NF;
    for (int s = tid; s < 4096; s += 256) {
      int t = s & 2047;
      int row = t >> 4, seg = t & 15;
      const _Float16* src = (s < 2048) ? s1 : s2;
      _Float16* dst = (s < 2048) ? buf0 : buf1;
      *(uint4*)&dst[row * PITCH + seg * 8] = *(const uint4*)&src[row * NF + seg * 8];
    }
  }
  __syncthreads();

  // per-lane x1 scalars (hoisted out of the o-loop)
  _Float16 x1v[64];   // x1[row = mb + 8*hl + r][col = nt*16 + ln]
  #pragma unroll
  for (int nt = 0; nt < 8; ++nt)
    #pragma unroll
    for (int r = 0; r < 8; ++r)
      x1v[nt * 8 + r] = buf0[(mb + hl * 8 + r) * PITCH + nt * 16 + ln];

  // A 16x32 f16 fragments from X2: lane holds row (mb+ln);
  // halves 0-7 = K(hl*8 ..), halves 8-15 = K(16+hl*8 ..)
  Frag16 afr[4];
  #pragma unroll
  for (int ks = 0; ks < 4; ++ks) {
    const _Float16* base = &buf1[(mb + ln) * PITCH + ks * 32 + hl * 8];
    afr[ks].q[0] = *(const uint4*)(base);
    afr[ks].q[1] = *(const uint4*)(base + 16);
  }
  __syncthreads();

  // ---- TDM double-buffered pipeline over output channels o ----
  const unsigned lds0 = (unsigned)(uintptr_t)buf0;
  const unsigned lds1 = (unsigned)(uintptr_t)buf1;
  if (w == 0) {
    tdm_load_tile(Wh + 0 * (size_t)(NF * NF), lds0);
    tdm_load_tile(Wh + 1 * (size_t)(NF * NF), lds1);
  }

  for (int o = 0; o < NF; ++o) {
    const _Float16* cw = (o & 1) ? buf1 : buf0;
    if (w == 0) {
      if (o < NF - 1) __builtin_amdgcn_s_wait_tensorcnt(1);  // load o complete (in-order)
      else            __builtin_amdgcn_s_wait_tensorcnt(0);
    }
    __syncthreads();   // W[o] visible to all waves

    float acc[8] = {0.f, 0.f, 0.f, 0.f, 0.f, 0.f, 0.f, 0.f};
    #pragma unroll
    for (int nt = 0; nt < 8; ++nt) {
      v8f c = {0.f, 0.f, 0.f, 0.f, 0.f, 0.f, 0.f, 0.f};
      #pragma unroll
      for (int ks = 0; ks < 4; ++ks) {
        // B 32x16 f16: lane holds col n = nt*16 + ln; halves 0-7 = K(ks*32 + hl*8 ..), halves 8-15 = K(ks*32 + 16 + hl*8 ..)
        Frag16 bfr;
        const _Float16* base = &cw[(nt * 16 + ln) * PITCH + ks * 32 + hl * 8];
        bfr.q[0] = *(const uint4*)(base);
        bfr.q[1] = *(const uint4*)(base + 16);
        c = __builtin_amdgcn_wmma_f32_16x16x32_f16(false, afr[ks].v, false, bfr.v,
                                                   (short)0, c, false, false);
      }
      // fused epilogue: acc[r] += T[row][i] * x1[row][i] for i = nt*16 + ln
      #pragma unroll
      for (int r = 0; r < 8; ++r)
        acc[r] += c[r] * (float)x1v[nt * 8 + r];
    }

    // reduce over the 16 lanes of each half (i dimension)
    #pragma unroll
    for (int m = 1; m <= 8; m <<= 1)
      #pragma unroll
      for (int r = 0; r < 8; ++r)
        acc[r] += __shfl_xor(acc[r], m, 32);

    // BN partial sums for this wave's 16 rows
    float sh = 0.f, qh = 0.f;
    #pragma unroll
    for (int r = 0; r < 8; ++r) { sh += acc[r]; qh += acc[r] * acc[r]; }

    // write y (Linear bias omitted: it cancels inside BatchNorm)
    #pragma unroll
    for (int r = 0; r < 8; ++r)
      if (ln == r)
        y[((size_t)(p * NB) + b0 + mb + hl * 8 + r) * NF + o] = acc[r];

    float sf = sh + __shfl_xor(sh, 16, 32);
    float qf = qh + __shfl_xor(qh, 16, 32);
    if (lane == 0) {
      int g = (p * NF + o) * (NTILES * 8) + bt * 8 + w;
      S[g] = sf;
      Q[g] = qf;
    }
    __syncthreads();   // everyone done with cw; safe for TDM to overwrite it

    if (w == 0 && o + 2 < NF)
      tdm_load_tile(Wh + (size_t)(o + 2) * (NF * NF), (o & 1) ? lds1 : lds0);
  }
}

// ---------------------------------------------------------------- per-(p,o) stats
__global__ void k_stats(const float* __restrict__ S, const float* __restrict__ Q,
                        float* __restrict__ meanA, float* __restrict__ invA) {
  __shared__ float ss[256], qq[256];
  const int g = blockIdx.x;          // 0..767 == p*128 + o
  const int tid = threadIdx.x;
  float s = S[g * 512 + tid] + S[g * 512 + 256 + tid];
  float q = Q[g * 512 + tid] + Q[g * 512 + 256 + tid];
  ss[tid] = s; qq[tid] = q;
  __syncthreads();
  for (int st = 128; st > 0; st >>= 1) {
    if (tid < st) { ss[tid] += ss[tid + st]; qq[tid] += qq[tid + st]; }
    __syncthreads();
  }
  if (tid == 0) {
    float mean = ss[0] * (1.0f / (float)NB);
    float var  = qq[0] * (1.0f / (float)NB) - mean * mean;  // biased, matches jnp.var
    meanA[g] = mean;
    invA[g]  = rsqrtf(var + 1e-5f);
  }
}

// ---------------------------------------------------------------- normalize + ReLU
__global__ void k_norm(const float* __restrict__ yws, const float* __restrict__ meanA,
                       const float* __restrict__ invA, const float* __restrict__ gamma,
                       const float* __restrict__ beta, float* __restrict__ out) {
  int idx = blockIdx.x * 256 + threadIdx.x;   // exactly 6,291,456 threads
  int o = idx & 127;
  int p = idx >> 20;                          // 8192*128 = 2^20
  int g = p * 128 + o;
  float v = yws[idx];
  float t = gamma[o] * (v - meanA[g]) * invA[g] + beta[o];
  out[idx] = t > 0.f ? t : 0.f;
}

// ---------------------------------------------------------------- launcher
extern "C" void kernel_launch(void* const* d_in, const int* in_sizes, int n_in,
                              void* d_out, int out_size, void* d_ws, size_t ws_size,
                              hipStream_t stream) {
  const float* x     = (const float*)d_in[0];   // [4,8192,128]
  const float* W     = (const float*)d_in[1];   // [128, 128*128]
  // d_in[2] = bias: cancels inside BatchNorm, unused
  const float* gamma = (const float*)d_in[3];
  const float* beta  = (const float*)d_in[4];

  char* ws = (char*)d_ws;
  _Float16* xh   = (_Float16*)(ws + XH_OFF);
  _Float16* Wh   = (_Float16*)(ws + WH_OFF);
  float* yws     = (float*)(ws + Y_OFF);
  float* S       = (float*)(ws + S_OFF);
  float* Q       = (float*)(ws + Q_OFF);
  float* meanA   = (float*)(ws + MEAN_OFF);
  float* invA    = (float*)(ws + INV_OFF);

  k_convert<<<24576, 256, 0, stream>>>(x, W, xh, Wh);

  dim3 grid(NTILES, NP);   // 64 batch tiles x 6 pairs
  size_t ldsBytes = 2u * WTILE * sizeof(_Float16);   // 69,632 B dynamic LDS
  k_bilinear<<<grid, 256, ldsBytes, stream>>>(xh, Wh, yws, S, Q);

  k_stats<<<NP * NF, 256, 0, stream>>>(S, Q, meanA, invA);

  k_norm<<<24576, 256, 0, stream>>>(yws, meanA, invA, gamma, beta, (float*)d_out);
}
